// MultiheadAttention_68719477030
// MI455X (gfx1250) — matrix operand order repaired and checked by reading, not compiled, not run on hardware
//
#include <hip/hip_runtime.h>
#include <math.h>

#define BATCH  2
#define SEQ    2048
#define DMODEL 1024
#define NHEAD  16
#define DHEAD  64
#define MROWS  (BATCH*SEQ)

typedef __bf16 v16bf __attribute__((ext_vector_type(16)));
typedef float  v8f   __attribute__((ext_vector_type(8)));

__device__ __forceinline__ v8f wmma_bf16(v16bf a, v16bf b, v8f c) {
  // D = A(16x32 bf16) * B(32x16 bf16) + C(16x16 f32)
  return __builtin_amdgcn_wmma_f32_16x16x32_bf16(false, a, false, b,
                                                 (short)0, c, false, false);
}

// 16-bit A-matrix 16x32 k-index pattern (contiguous, matches B):
// lane L: m = L%16, hi = L/16; VGPR j holds k, k+1 with
//   k = 16*hi + 2j
__device__ __forceinline__ int kpatA(int j, int hi) {
  return 16*hi + 2*j;
}

__device__ __forceinline__ v16bf a_frag_bf16(const __bf16* p, int hi) {
  v16bf a;
#pragma unroll
  for (int j = 0; j < 8; ++j) {
    const int k = kpatA(j, hi);
    a[2*j]   = p[k];
    a[2*j+1] = p[k+1];
  }
  return a;
}
__device__ __forceinline__ v16bf a_frag_f32(const float* p, int hi) {
  v16bf a;
#pragma unroll
  for (int j = 0; j < 8; ++j) {
    const int k = kpatA(j, hi);
    a[2*j]   = (__bf16)p[k];
    a[2*j+1] = (__bf16)p[k+1];
  }
  return a;
}
// 16-bit B fragment 32x16 (contiguous k, matches A): lane L: n = L%16, hi = L/16; VGPR j: k = 16*hi+2j
__device__ __forceinline__ v16bf b_frag_bf16(const __bf16* p, int hi) {
  v16bf b;
#pragma unroll
  for (int j = 0; j < 8; ++j) {
    const int k = 16*hi + 2*j;
    b[2*j]   = p[k];
    b[2*j+1] = p[k+1];
  }
  return b;
}

// gfx1250 async global->LDS copy, 16 bytes per lane (tracked by ASYNCcnt).
__device__ __forceinline__ void async_b128(unsigned lds_off, const void* g) {
  unsigned long long ga = (unsigned long long)(size_t)g;
  asm volatile("global_load_async_to_lds_b128 %0, %1, off"
               :: "v"(lds_off), "v"(ga) : "memory");
}
__device__ __forceinline__ unsigned lds_off_of(const void* p) {
  return (unsigned)(size_t)p;   // LDS aperture: low 32 bits == LDS byte offset
}

// ---------------------------------------------------------------------
// f32 -> bf16 pre-conversion (weights), 4 floats per thread
// ---------------------------------------------------------------------
__global__ __launch_bounds__(256)
void cvt_f32_bf16(const float* __restrict__ src, __bf16* __restrict__ dst, int n) {
  const int i = (blockIdx.x * 256 + threadIdx.x) * 4;
  if (i + 3 < n) {
    const float4 v = *(const float4*)(src + i);
    dst[i]   = (__bf16)v.x;
    dst[i+1] = (__bf16)v.y;
    dst[i+2] = (__bf16)v.z;
    dst[i+3] = (__bf16)v.w;
  }
}

// =====================================================================
// GEMM: C[m,n] = sum_k A[m,k] * Wb[n,k] + bias[n]   (torch Linear)
// 8 waves / block; block tile 128(M) x 64(N); W tile (64x32 bf16, 4 KB)
// double-buffered in LDS via global_load_async_to_lds_b128.
// OMODE 0: bf16 [M,N]  |  1: bf16 Vt[b,h,d,s]  |  2: f32 [M,N]
// =====================================================================
template<int OMODE, bool ABF16>
__global__ __launch_bounds__(256)
void gemm_xwT(const void* __restrict__ Ap, const __bf16* __restrict__ Wb,
              const float* __restrict__ bias, void* __restrict__ Out)
{
  __shared__ __bf16 Wt[2][64][32];            // [buf][n_local][k_local]
  const int tid  = threadIdx.x;
  const int lane = tid & 31;
  const int wv   = tid >> 5;                  // 0..7
  const int hi   = lane >> 4, ln = lane & 15;
  const int m0   = blockIdx.x * 128 + wv * 16;
  const int n0   = blockIdx.y * 64;

  const int crow = tid >> 2;                  // 0..63 : n_local
  const int ccol = (tid & 3) * 8;             // 0,8,16,24 : k_local (8 bf16 = 16B)

  const float*  Af = (const float*)Ap;
  const __bf16* Ab = (const __bf16*)Ap;

  // prologue: stage k0 = 0 into buffer 0
  async_b128(lds_off_of(&Wt[0][crow][ccol]),
             Wb + (size_t)(n0 + crow) * DMODEL + ccol);

  v8f acc[4] = {};
  const int NK = DMODEL / 32;                 // 32 k-steps
  for (int it = 0; it < NK; ++it) {
    const int k0  = it * 32;
    const int cur = it & 1;
    if (it + 1 < NK) {
      async_b128(lds_off_of(&Wt[cur ^ 1][crow][ccol]),
                 Wb + (size_t)(n0 + crow) * DMODEL + (k0 + 32) + ccol);
      asm volatile("s_wait_asynccnt 0x1" ::: "memory");  // prev tile landed
    } else {
      asm volatile("s_wait_asynccnt 0x0" ::: "memory");
    }
    __syncthreads();

    v16bf a;
    if constexpr (ABF16) {
      const __bf16* arow = Ab + (size_t)(m0 + ln) * DMODEL + k0;
      a = a_frag_bf16(arow, hi);
      if (k0 + 32 < DMODEL) __builtin_prefetch(arow + 32, 0, 1);
    } else {
      const float* arow = Af + (size_t)(m0 + ln) * DMODEL + k0;
      a = a_frag_f32(arow, hi);
      if (k0 + 32 < DMODEL) __builtin_prefetch(arow + 32, 0, 1);
    }

#pragma unroll
    for (int nt = 0; nt < 4; ++nt) {
      v16bf bf;
#pragma unroll
      for (int j = 0; j < 8; ++j) {
        const int k = 16*hi + 2*j;
        bf[2*j]   = Wt[cur][nt*16 + ln][k];
        bf[2*j+1] = Wt[cur][nt*16 + ln][k + 1];
      }
      acc[nt] = wmma_bf16(a, bf, acc[nt]);
    }
    __syncthreads();                          // all waves done with buf `cur`
  }

  // C/D layout: lane L -> column n = L%16 ; VGPR r -> row m = 8*(L/16)+r
#pragma unroll
  for (int nt = 0; nt < 4; ++nt) {
    const int n  = n0 + nt*16 + ln;
    const float bs = bias[n];
    if constexpr (OMODE == 0) {
      __bf16* Ob = (__bf16*)Out;
#pragma unroll
      for (int r = 0; r < 8; ++r) {
        const int m = m0 + 8*hi + r;
        Ob[(size_t)m * DMODEL + n] = (__bf16)(acc[nt][r] + bs);
      }
    } else if constexpr (OMODE == 1) {
      __bf16* Ob = (__bf16*)Out;               // Vt[b,h,d,s]
      const int h  = n / DHEAD;
      const int dl = n % DHEAD;
      const int b  = m0 / SEQ;                 // 128-row tile never crosses batch
      const int s0 = m0 % SEQ + 8*hi;
      __bf16* dst = Ob + (((size_t)(b*NHEAD + h))*DHEAD + dl) * SEQ + s0;
#pragma unroll
      for (int r = 0; r < 8; ++r)
        dst[r] = (__bf16)(acc[nt][r] + bs);    // contiguous in s
    } else {
      float* Of = (float*)Out;
#pragma unroll
      for (int r = 0; r < 8; ++r) {
        const int m = m0 + 8*hi + r;
        Of[(size_t)m * DMODEL + n] = acc[nt][r] + bs;
      }
    }
  }
}

// =====================================================================
// Flash attention: 8 waves / block = 8 query tiles of one (b,h) sharing
// async-LDS double-buffered K (32x64) and V^T (64x32) tiles per 32-key
// chunk. scores^T = K·Q^T (softmax lane-local per query column);
// context^T accumulated with online-softmax rescaling.
// =====================================================================
__global__ __launch_bounds__(256)
void attn_kernel(const __bf16* __restrict__ Qb, const __bf16* __restrict__ Kb,
                 const __bf16* __restrict__ VtG, const unsigned char* __restrict__ mask,
                 __bf16* __restrict__ Ctx)
{
  __shared__ __bf16 Kt[2][32][64];            // [buf][key_local][d]
  __shared__ __bf16 Vs[2][64][32];            // [buf][d_local][s_local]

  const int tid  = threadIdx.x;
  const int lane = tid & 31;
  const int wv   = tid >> 5;
  const int hi   = lane >> 4, ln = lane & 15;
  const int qt   = blockIdx.x * 8 + wv;
  const int h    = blockIdx.y;
  const int b    = blockIdx.z;
  const float scale = 0.125f;                 // 1/sqrt(64)
  const float NEG_INF = -__builtin_inff();

  // async-copy coords
  const int krow = tid >> 3, kcol = (tid & 7) * 8;   // Kt: 32 rows x 128B
  const int vrow = tid >> 2, vcol = (tid & 3) * 8;   // Vs: 64 rows x 64B
  const __bf16* Kg = Kb  + (size_t)(b*SEQ) * DMODEL + h*DHEAD;
  const __bf16* Vg = VtG + ((size_t)(b*NHEAD + h)) * DHEAD * SEQ;

  // Q^T B-fragments (k-dim = head dim, contiguous per query row)
  const __bf16* qrow = Qb + (size_t)(b*SEQ + qt*16 + ln) * DMODEL + h*DHEAD;
  const v16bf bq0 = b_frag_bf16(qrow,      hi);   // d = 0..31
  const v16bf bq1 = b_frag_bf16(qrow + 32, hi);   // d = 32..63

  // prologue: stage chunk 0
  async_b128(lds_off_of(&Kt[0][krow][kcol]), Kg + (size_t)krow * DMODEL + kcol);
  async_b128(lds_off_of(&Vs[0][vrow][vcol]), Vg + (size_t)vrow * SEQ + vcol);

  v8f   ctx[4] = {};
  float m_run = NEG_INF;
  float l_run = 0.0f;

  const int NC = SEQ / 32;                    // 64 chunks
  for (int it = 0; it < NC; ++it) {
    const int s0  = it * 32;
    const int cur = it & 1;
    if (it + 1 < NC) {
      async_b128(lds_off_of(&Kt[cur ^ 1][krow][kcol]),
                 Kg + (size_t)(s0 + 32 + krow) * DMODEL + kcol);
      async_b128(lds_off_of(&Vs[cur ^ 1][vrow][vcol]),
                 Vg + (size_t)vrow * SEQ + (s0 + 32) + vcol);
      asm volatile("s_wait_asynccnt 0x2" ::: "memory");  // prev pair landed
    } else {
      asm volatile("s_wait_asynccnt 0x0" ::: "memory");
    }
    __syncthreads();

    // two 16-key score^T fragments from LDS K tile
    v8f sc[2];
#pragma unroll
    for (int c = 0; c < 2; ++c) {
      v16bf aK0, aK1;
#pragma unroll
      for (int j = 0; j < 8; ++j) {
        const int k = kpatA(j, hi);
        aK0[2*j]   = Kt[cur][c*16 + ln][k];
        aK0[2*j+1] = Kt[cur][c*16 + ln][k + 1];
        aK1[2*j]   = Kt[cur][c*16 + ln][32 + k];
        aK1[2*j+1] = Kt[cur][c*16 + ln][32 + k + 1];
      }
      v8f z = {};
      z = wmma_bf16(aK0, bq0, z);
      z = wmma_bf16(aK1, bq1, z);
      sc[c] = z;
    }
    // scale + mask (mask shape [B,1,1,S] over keys)
#pragma unroll
    for (int c = 0; c < 2; ++c)
#pragma unroll
      for (int r = 0; r < 8; ++r) {
        const int key = s0 + c*16 + 8*hi + r;
        const float v = sc[c][r] * scale;
        sc[c][r] = mask[(size_t)b*SEQ + key] ? v : NEG_INF;
      }
    // swap half-waves so each lane owns 16 consecutive keys for its query t
    float arr[16];
#pragma unroll
    for (int r = 0; r < 8; ++r) {
      const float p0 = __shfl_xor(sc[0][r], 16, 32);
      const float p1 = __shfl_xor(sc[1][r], 16, 32);
      arr[r]     = hi ? p1       : sc[0][r];   // keys s0+16*hi + 0..7
      arr[8 + r] = hi ? sc[1][r] : p0;         // keys s0+16*hi + 8..15
    }
    float mx = NEG_INF;
#pragma unroll
    for (int i = 0; i < 16; ++i) mx = fmaxf(mx, arr[i]);
    mx = fmaxf(mx, __shfl_xor(mx, 16, 32));    // max over all 32 keys of chunk
    const float m_new = fmaxf(m_run, mx);
    const float alpha = (m_new == NEG_INF) ? 0.0f : __expf(m_run - m_new);
    float sum = 0.0f;
#pragma unroll
    for (int i = 0; i < 16; ++i) {
      const float e = (arr[i] == NEG_INF) ? 0.0f : __expf(arr[i] - m_new);
      arr[i] = e;
      sum += e;
    }
    l_run = l_run * alpha + sum;
    m_run = m_new;
#pragma unroll
    for (int dt = 0; dt < 4; ++dt)
#pragma unroll
      for (int r = 0; r < 8; ++r) ctx[dt][r] *= alpha;

    // pack P^T B-fragment: lane needs k = 16*hi + i, exactly arr[i]
    v16bf pb;
#pragma unroll
    for (int i = 0; i < 16; ++i) pb[i] = (__bf16)arr[i];

    // context^T += V^T(16x32) * P^T(32x16), 4 d-tiles, from LDS V tile
#pragma unroll
    for (int dt = 0; dt < 4; ++dt) {
      v16bf av;
#pragma unroll
      for (int j = 0; j < 8; ++j) {
        const int k = kpatA(j, hi);
        av[2*j]   = Vs[cur][dt*16 + ln][k];
        av[2*j+1] = Vs[cur][dt*16 + ln][k + 1];
      }
      ctx[dt] = wmma_bf16(av, pb, ctx[dt]);
    }
    __syncthreads();                           // all waves done with buf `cur`
  }

  const float l_tot = l_run + __shfl_xor(l_run, 16, 32);
  const float inv = (l_tot > 0.0f) ? 1.0f / l_tot : 0.0f;

  // Ctx[b, qt*16+t, h*DHEAD + d] ; lane column t = ln, rows d = dt*16+8*hi+r
  __bf16* orow = Ctx + (size_t)(b*SEQ + qt*16 + ln) * DMODEL + h*DHEAD;
#pragma unroll
  for (int dt = 0; dt < 4; ++dt)
#pragma unroll
    for (int r = 0; r < 8; ++r)
      orow[dt*16 + 8*hi + r] = (__bf16)(ctx[dt][r] * inv);
}

extern "C" void kernel_launch(void* const* d_in, const int* in_sizes, int n_in,
                              void* d_out, int out_size, void* d_ws, size_t ws_size,
                              hipStream_t stream) {
  const float* query = (const float*)d_in[0];
  const float* key   = (const float*)d_in[1];
  const float* value = (const float*)d_in[2];
  const unsigned char* mask = (const unsigned char*)d_in[3];
  const float* wq = (const float*)d_in[4];
  const float* bq = (const float*)d_in[5];
  const float* wk = (const float*)d_in[6];
  const float* bk = (const float*)d_in[7];
  const float* wv = (const float*)d_in[8];
  const float* bv = (const float*)d_in[9];
  const float* wo = (const float*)d_in[10];
  const float* bo = (const float*)d_in[11];

  const size_t nBSD = (size_t)MROWS * DMODEL;     // 4M elements
  const size_t nDD  = (size_t)DMODEL * DMODEL;    // 1M elements
  __bf16* Qb  = (__bf16*)d_ws;                    //  8 MiB
  __bf16* Kb  = Qb + nBSD;                        //  8 MiB
  __bf16* Vt  = Kb + nBSD;                        //  8 MiB  [B,H,DHEAD,S]
  __bf16* Ctx = Vt + nBSD;                        //  8 MiB
  __bf16* Wqb = Ctx + nBSD;                       //  2 MiB each
  __bf16* Wkb = Wqb + nDD;
  __bf16* Wvb = Wkb + nDD;
  __bf16* Wob = Wvb + nDD;                        // 40 MiB total

  const dim3 blk(256);
  const dim3 gc((unsigned)(nDD / (256 * 4)));
  cvt_f32_bf16<<<gc, blk, 0, stream>>>(wq, Wqb, (int)nDD);
  cvt_f32_bf16<<<gc, blk, 0, stream>>>(wk, Wkb, (int)nDD);
  cvt_f32_bf16<<<gc, blk, 0, stream>>>(wv, Wvb, (int)nDD);
  cvt_f32_bf16<<<gc, blk, 0, stream>>>(wo, Wob, (int)nDD);

  const dim3 gg(MROWS / 128, DMODEL / 64);        // 32 x 16 blocks, 8 waves each
  gemm_xwT<0, false><<<gg, blk, 0, stream>>>((const void*)query, Wqb, bq, (void*)Qb);
  gemm_xwT<0, false><<<gg, blk, 0, stream>>>((const void*)key,   Wkb, bk, (void*)Kb);
  gemm_xwT<1, false><<<gg, blk, 0, stream>>>((const void*)value, Wvb, bv, (void*)Vt);

  const dim3 ga(SEQ / 16 / 8, NHEAD, BATCH);      // 16 x 16 x 2 blocks
  attn_kernel<<<ga, blk, 0, stream>>>(Qb, Kb, Vt, mask, Ctx);

  gemm_xwT<2, true><<<gg, blk, 0, stream>>>((const void*)Ctx, Wob, bo, d_out);
}
